// Retraction_49271864820311
// MI455X (gfx1250) — compile-verified, matrix operand order checked by reading, not run on hardware
//
#include <hip/hip_runtime.h>
#include <hip/hip_bf16.h>

#define NPROB 8192
#define DD 256
#define KK 32
#define NS_ITERS 12

typedef __attribute__((ext_vector_type(16))) __bf16 v16bf;
typedef __attribute__((ext_vector_type(8)))  float  v8f;

// ---- LDS layout (bytes) ----
#define OFF_PVT_HI 0          // bf16 [32][256]  PV^T hi
#define OFF_PVT_LO 16384      // bf16 [32][256]  PV^T lo
#define OFF_PV_HI  32768      // bf16 [256][32]  PV hi
#define OFF_PV_LO  49152      // bf16 [256][32]  PV lo
#define OFF_GPART  65536      // f32  [2][32][32] partial Gram
#define OFF_ZF     73728      // f32  [32][32]   Z (f32 master copy)
#define OFF_TF     77824      // f32  [32][32]   scratch (T / Ynew)
#define OFF_T2F    81920      // f32  [32][32]   scratch (Znew)
#define OFF_YH     86016      // bf16 [32][32]
#define OFF_YL     88064
#define OFF_ZH     90112
#define OFF_ZL     92160
#define OFF_WH     94208
#define OFF_WL     96256
#define OFF_RED    98304      // f32  [256]
#define SMEM_BYTES 99328

__device__ __forceinline__ void bf16_split(float x, __bf16& hi, __bf16& lo) {
    hi = (__bf16)x;
    lo = (__bf16)(x - (float)hi);
}

// f32-accurate bf16 WMMA: hi*hi + hi*lo + lo*hi, f32 accumulate
__device__ __forceinline__ v8f mm3(v16bf a_hi, v16bf a_lo, v16bf b_hi, v16bf b_lo, v8f acc) {
    acc = __builtin_amdgcn_wmma_f32_16x16x32_bf16(false, a_hi, false, b_hi, (short)0, acc, false, false);
    acc = __builtin_amdgcn_wmma_f32_16x16x32_bf16(false, a_hi, false, b_lo, (short)0, acc, false, false);
    acc = __builtin_amdgcn_wmma_f32_16x16x32_bf16(false, a_lo, false, b_hi, (short)0, acc, false, false);
    return acc;
}

__global__ __launch_bounds__(256)
void retraction_polar_kernel(const float* __restrict__ inp,
                             const float* __restrict__ grd,
                             const float* __restrict__ lrp,
                             float* __restrict__ out)
{
    extern __shared__ char smem[];
    __bf16* PVT_hi = (__bf16*)(smem + OFF_PVT_HI);
    __bf16* PVT_lo = (__bf16*)(smem + OFF_PVT_LO);
    __bf16* PV_hi  = (__bf16*)(smem + OFF_PV_HI);
    __bf16* PV_lo  = (__bf16*)(smem + OFF_PV_LO);
    float*  Gpart  = (float*) (smem + OFF_GPART);
    float*  Zf     = (float*) (smem + OFF_ZF);
    float*  Tf     = (float*) (smem + OFF_TF);
    float*  T2f    = (float*) (smem + OFF_T2F);
    __bf16* Yh     = (__bf16*)(smem + OFF_YH);
    __bf16* Yl     = (__bf16*)(smem + OFF_YL);
    __bf16* Zh     = (__bf16*)(smem + OFF_ZH);
    __bf16* Zl     = (__bf16*)(smem + OFF_ZL);
    __bf16* Wh     = (__bf16*)(smem + OFF_WH);
    __bf16* Wl     = (__bf16*)(smem + OFF_WL);
    float*  red    = (float*) (smem + OFF_RED);

    const int n    = blockIdx.x;
    const int t    = threadIdx.x;
    const int wave = t >> 5;
    const int lane = t & 31;
    const float lr = *lrp;

    const float* inN = inp + (size_t)n * (DD * KK);
    const float* grN = grd + (size_t)n * (DD * KK);
    float*      outN = out + (size_t)n * (DD * KK);

    // ---- Stage 1: PV = in - lr*g, bf16 hi/lo split, stage row-major + transposed ----
    #pragma unroll
    for (int it = 0; it < 8; ++it) {
        int l = it * 1024 + t * 4;          // 16B-aligned
        int d = l >> 5, k = l & 31;
        float4 a = *(const float4*)(inN + l);
        float4 g = *(const float4*)(grN + l);
        float pv[4] = { a.x - lr * g.x, a.y - lr * g.y, a.z - lr * g.z, a.w - lr * g.w };
        #pragma unroll
        for (int c = 0; c < 4; ++c) {
            __bf16 hi, lo;
            bf16_split(pv[c], hi, lo);
            PV_hi [d * 32 + k + c]   = hi;
            PV_lo [d * 32 + k + c]   = lo;
            PVT_hi[(k + c) * 256 + d] = hi;   // transposed copy for Gram A/B tiles
            PVT_lo[(k + c) * 256 + d] = lo;
        }
    }
    __syncthreads();

    // ---- Stage 2: G = PV^T PV.  8 waves = 4 output tiles x 2 D-halves.       ----
    // A tile (16x32) and B tile (32x16) both read as contiguous rows of PV^T.
    {
        const int ti = (wave >> 2) & 1, tj = (wave >> 1) & 1, h = wave & 1;
        const int m = lane & 15, half = lane >> 4;
        v8f acc = {};
        #pragma unroll
        for (int kc = 0; kc < 4; ++kc) {
            int koff = (h * 4 + kc) * 32 + half * 16;   // 32B-aligned chunk of a PV^T row
            v16bf a_hi = *(const v16bf*)(PVT_hi + (ti * 16 + m) * 256 + koff);
            v16bf a_lo = *(const v16bf*)(PVT_lo + (ti * 16 + m) * 256 + koff);
            v16bf b_hi = *(const v16bf*)(PVT_hi + (tj * 16 + m) * 256 + koff);
            v16bf b_lo = *(const v16bf*)(PVT_lo + (tj * 16 + m) * 256 + koff);
            acc = mm3(a_hi, a_lo, b_hi, b_lo, acc);
        }
        float* gp = Gpart + h * 1024;
        #pragma unroll
        for (int r = 0; r < 8; ++r)
            gp[(ti * 16 + r + half * 8) * 32 + tj * 16 + m] = acc[r];
    }
    __syncthreads();

    // ---- Stage 3: reduce halves, Frobenius norm, init Y = G/||G||_F, Z = I ----
    float gv[4], sq = 0.f;
    #pragma unroll
    for (int j = 0; j < 4; ++j) {
        int i = t + j * 256;
        float v = Gpart[i] + Gpart[i + 1024];
        gv[j] = v;
        sq += v * v;
    }
    red[t] = sq;
    __syncthreads();
    #pragma unroll
    for (int s = 128; s > 0; s >>= 1) {
        if (t < s) red[t] += red[t + s];
        __syncthreads();
    }
    const float cnorm = sqrtf(red[0]);
    const float inv_c = 1.f / cnorm;

    #pragma unroll
    for (int j = 0; j < 4; ++j) {
        int i = t + j * 256;
        int row = i >> 5, col = i & 31;
        float y = gv[j] * inv_c;
        __bf16 hi, lo;
        bf16_split(y, hi, lo);
        Yh[i] = hi; Yl[i] = lo;
        float z = (row == col) ? 1.f : 0.f;
        Zf[i] = z;
        Zh[i] = (__bf16)z; Zl[i] = (__bf16)0.f;
    }
    __syncthreads();

    // ---- Stage 4: Newton-Schulz for A^{-1/2}:                                ----
    //   T = Z*Y ; W = 1.5 I - 0.5 T ; Y <- Y*W ; Z <- W*Z
    const int ti = (wave >> 1) & 1, tj = wave & 1;   // tile for 32x32 matmuls
    const int m = lane & 15, half = lane >> 4;
    for (int iter = 0; iter < NS_ITERS; ++iter) {
        if (wave < 4) {                               // T = Z*Y  (wave-uniform branch, EXEC all-ones)
            v8f acc = {};
            v16bf a_hi = *(const v16bf*)(Zh + (ti * 16 + m) * 32 + half * 16);
            v16bf a_lo = *(const v16bf*)(Zl + (ti * 16 + m) * 32 + half * 16);
            v16bf b_hi = *(const v16bf*)(Yh + (tj * 16 + m) * 32 + half * 16);  // Y symmetric: row == column
            v16bf b_lo = *(const v16bf*)(Yl + (tj * 16 + m) * 32 + half * 16);
            acc = mm3(a_hi, a_lo, b_hi, b_lo, acc);
            #pragma unroll
            for (int r = 0; r < 8; ++r)
                Tf[(ti * 16 + r + half * 8) * 32 + tj * 16 + m] = acc[r];
        }
        __syncthreads();
        #pragma unroll
        for (int j = 0; j < 4; ++j) {                 // W = 1.5I - 0.5T
            int i = t + j * 256;
            int row = i >> 5, col = i & 31;
            float w = ((row == col) ? 1.5f : 0.f) - 0.5f * Tf[i];
            __bf16 hi, lo;
            bf16_split(w, hi, lo);
            Wh[i] = hi; Wl[i] = lo;
        }
        __syncthreads();
        {                                             // waves 0-3: Ynew = Y*W ; waves 4-7: Znew = W*Z
            const __bf16* Ah = (wave < 4) ? Yh : Wh;
            const __bf16* Al = (wave < 4) ? Yl : Wl;
            const __bf16* Bh = (wave < 4) ? Wh : Zh;
            const __bf16* Bl = (wave < 4) ? Wl : Zl;
            float*       Dst = (wave < 4) ? Tf : T2f;
            v8f acc = {};
            v16bf a_hi = *(const v16bf*)(Ah + (ti * 16 + m) * 32 + half * 16);
            v16bf a_lo = *(const v16bf*)(Al + (ti * 16 + m) * 32 + half * 16);
            v16bf b_hi = *(const v16bf*)(Bh + (tj * 16 + m) * 32 + half * 16);
            v16bf b_lo = *(const v16bf*)(Bl + (tj * 16 + m) * 32 + half * 16);
            acc = mm3(a_hi, a_lo, b_hi, b_lo, acc);
            #pragma unroll
            for (int r = 0; r < 8; ++r)
                Dst[(ti * 16 + r + half * 8) * 32 + tj * 16 + m] = acc[r];
        }
        __syncthreads();
        #pragma unroll
        for (int j = 0; j < 4; ++j) {                 // commit Y, Z (+ bf16 splits)
            int i = t + j * 256;
            float y = Tf[i];
            __bf16 hi, lo;
            bf16_split(y, hi, lo);
            Yh[i] = hi; Yl[i] = lo;
            float z = T2f[i];
            Zf[i] = z;
            bf16_split(z, hi, lo);
            Zh[i] = hi; Zl[i] = lo;
        }
        __syncthreads();
    }

    // ---- Stage 5: P = G^{-1/2} = Z / sqrt(c); reuse W buffers for P hi/lo ----
    {
        const float isc = rsqrtf(cnorm);
        #pragma unroll
        for (int j = 0; j < 4; ++j) {
            int i = t + j * 256;
            float p = Zf[i] * isc;
            __bf16 hi, lo;
            bf16_split(p, hi, lo);
            Wh[i] = hi; Wl[i] = lo;
        }
    }
    __syncthreads();

    // ---- Stage 6: out = PV @ P.  16 d-tiles x 2 k-tiles over 8 waves.       ----
    // P symmetric => B column n == P row n (contiguous LDS read).
    #pragma unroll
    for (int dt = wave * 2; dt < wave * 2 + 2; ++dt) {
        #pragma unroll
        for (int cj = 0; cj < 2; ++cj) {
            v8f acc = {};
            v16bf a_hi = *(const v16bf*)(PV_hi + (dt * 16 + m) * 32 + half * 16);
            v16bf a_lo = *(const v16bf*)(PV_lo + (dt * 16 + m) * 32 + half * 16);
            v16bf b_hi = *(const v16bf*)(Wh + (cj * 16 + m) * 32 + half * 16);
            v16bf b_lo = *(const v16bf*)(Wl + (cj * 16 + m) * 32 + half * 16);
            acc = mm3(a_hi, a_lo, b_hi, b_lo, acc);
            #pragma unroll
            for (int r = 0; r < 8; ++r)
                outN[(dt * 16 + r + half * 8) * 32 + cj * 16 + m] = acc[r];
        }
    }
}

extern "C" void kernel_launch(void* const* d_in, const int* in_sizes, int n_in,
                              void* d_out, int out_size, void* d_ws, size_t ws_size,
                              hipStream_t stream) {
    (void)in_sizes; (void)n_in; (void)out_size; (void)d_ws; (void)ws_size;
    const float* inp = (const float*)d_in[0];
    const float* grd = (const float*)d_in[1];
    const float* lrp = (const float*)d_in[2];
    float* out = (float*)d_out;
    retraction_polar_kernel<<<NPROB, 256, SMEM_BYTES, stream>>>(inp, grd, lrp, out);
}
